// SVDTPAAttention_23184233464210
// MI455X (gfx1250) — matrix operand order repaired and checked by reading, not compiled, not run on hardware
//
#include <hip/hip_runtime.h>

// ---------------- types ----------------
typedef __attribute__((ext_vector_type(16))) __bf16 v16bf;
typedef __attribute__((ext_vector_type(8)))  __bf16 v8bf;
typedef __attribute__((ext_vector_type(8)))  float  v8f;

// ---------------- problem constants ----------------
#define B_    2
#define S_    2048
#define HID_  2048
#define H_    16
#define HKV_  8
#define D_    128
#define R_    64
#define BS_   4096            // B_*S_
#define SCALE_ 0.08838834764831845f   // D^-0.5

static __device__ __forceinline__ v8f wmma_bf16(v16bf a, v16bf b, v8f c) {
  // D = A(16x32 bf16) * B(32x16 bf16) + C(16x16 f32)
  return __builtin_amdgcn_wmma_f32_16x16x32_bf16(false, a, false, b, (short)0, c, false, false);
}

static __device__ __forceinline__ float half16_max(float v) {
  v = fmaxf(v, __shfl_xor(v, 1, 32));
  v = fmaxf(v, __shfl_xor(v, 2, 32));
  v = fmaxf(v, __shfl_xor(v, 4, 32));
  v = fmaxf(v, __shfl_xor(v, 8, 32));
  return v;
}
static __device__ __forceinline__ float half16_sum(float v) {
  v += __shfl_xor(v, 1, 32);
  v += __shfl_xor(v, 2, 32);
  v += __shfl_xor(v, 4, 32);
  v += __shfl_xor(v, 8, 32);
  return v;
}

// ---------------- elementwise kernels ----------------
__global__ __launch_bounds__(256) void cvt_f32_bf16(const float* __restrict__ src,
                                                    __bf16* __restrict__ dst, long n) {
  long i = (long)blockIdx.x * 256 + threadIdx.x;
  if (i < n) dst[i] = (__bf16)src[i];
}

// B_x[h][r][d] (f32) -> out[h][d][r] (bf16)
__global__ __launch_bounds__(256) void transpose_rd(const float* __restrict__ src,
                                                    __bf16* __restrict__ dst, int Hn) {
  long i = (long)blockIdx.x * 256 + threadIdx.x;
  long n = (long)Hn * D_ * R_;
  if (i >= n) return;
  int r = (int)(i % R_);
  int d = (int)((i / R_) % D_);
  int h = (int)(i / ((long)R_ * D_));
  dst[i] = (__bf16)src[((long)h * R_ + r) * D_ + d];
}

// qk rope: lin[b][s][heads*128] f32 -> out[(b*heads+h)][s][128] bf16, scaled
__global__ __launch_bounds__(256) void rope_kernel(const float* __restrict__ lin,
                                                   const float* __restrict__ fc,
                                                   const float* __restrict__ fs,
                                                   __bf16* __restrict__ out,
                                                   int heads, float scale) {
  long i = (long)blockIdx.x * 256 + threadIdx.x;
  long n = (long)BS_ * heads * 64;
  if (i >= n) return;
  int d2 = (int)(i & 63);
  long t = i >> 6;
  int h = (int)(t % heads); t /= heads;
  int s = (int)(t % S_);
  int b = (int)(t / S_);
  const float* base = lin + (((long)b * S_ + s) * heads + h) * D_;
  float x1 = base[d2], x2 = base[d2 + 64];
  float c = fc[(long)s * 64 + d2], sn = fs[(long)s * 64 + d2];
  __bf16* ob = out + (((long)(b * heads + h)) * S_ + s) * D_;
  ob[d2]      = (__bf16)((x1 * c - x2 * sn) * scale);
  ob[d2 + 64] = (__bf16)((x1 * sn + x2 * c) * scale);
}

// vlin[b][s][HKV*128] f32 -> vT[(b*HKV+hk)][d][s] bf16
__global__ __launch_bounds__(256) void vtrans_kernel(const float* __restrict__ vlin,
                                                     __bf16* __restrict__ vT) {
  long i = (long)blockIdx.x * 256 + threadIdx.x;
  long n = (long)B_ * HKV_ * D_ * S_;
  if (i >= n) return;
  int s  = (int)(i % S_);
  int d  = (int)((i / S_) % D_);
  long t = i / ((long)S_ * D_);
  int hk = (int)(t % HKV_);
  int b  = (int)(t / HKV_);
  vT[i] = (__bf16)vlin[(((long)b * S_ + s) * HKV_ + hk) * D_ + d];
}

// ---------------- generic C = scale * (A[M,K] x B[N,K]^T) bf16 GEMM ----------------
// One wave -> 32x64 output tile (2 M-subtiles x 4 N-subtiles); B fragments are
// reused across both M-subtiles -> 12 b128 loads per 8 WMMAs per K-step.
// grid = (N/64, M/128, batch), block = 128 (4 waves).
template <bool OUT_BF16>
__global__ __launch_bounds__(128) void gemm_bt_kernel(
    const __bf16* __restrict__ A, int lda, long aoff,
    const __bf16* __restrict__ Bm, int ldb, long boff,
    void* __restrict__ Cv, int ldc, long coff,
    int M, int N, int K, float scale) {
  int batch = blockIdx.z;
  const __bf16* Ab = A + (long)batch * aoff;
  const __bf16* Bb = Bm + (long)batch * boff;
  int wave = threadIdx.x >> 5;
  int lane = threadIdx.x & 31;
  int l16 = lane & 15, hi = lane >> 4;
  int mt = blockIdx.y * 4 + wave;  // 32-row tile index
  int nt = blockIdx.x;             // 64-col tile index
  if (mt * 32 >= M) return;

  v8f acc[2][4] = {};
  long arow0 = (long)(mt * 32 + l16) * lda;
  long arow1 = (long)(mt * 32 + 16 + l16) * lda;
  for (int kk = 0; kk < K; kk += 32) {
    // A fragments: lane=row; low half holds K 0..15, high half K 16..31 (contiguous)
    v16bf a0, a1;
    {
      v8bf lo = *(const v8bf*)(Ab + arow0 + kk + hi * 16);
      v8bf hh = *(const v8bf*)(Ab + arow0 + kk + 8 + hi * 16);
#pragma unroll
      for (int i = 0; i < 8; ++i) { a0[i] = lo[i]; a0[8 + i] = hh[i]; }
    }
    {
      v8bf lo = *(const v8bf*)(Ab + arow1 + kk + hi * 16);
      v8bf hh = *(const v8bf*)(Ab + arow1 + kk + 8 + hi * 16);
#pragma unroll
      for (int i = 0; i < 8; ++i) { a1[i] = lo[i]; a1[8 + i] = hh[i]; }
    }
#pragma unroll
    for (int j = 0; j < 4; ++j) {
      // B fragment: lane=column; low half K 0..15, high half K 16..31 (contiguous)
      long brow = (long)(nt * 64 + j * 16 + l16) * ldb;
      v16bf bb = *(const v16bf*)(Bb + brow + kk + hi * 16);
      acc[0][j] = wmma_bf16(a0, bb, acc[0][j]);
      acc[1][j] = wmma_bf16(a1, bb, acc[1][j]);
    }
  }
#pragma unroll
  for (int mi = 0; mi < 2; ++mi) {
#pragma unroll
    for (int j = 0; j < 4; ++j) {
#pragma unroll
      for (int r = 0; r < 8; ++r) {
        int mrow = mt * 32 + mi * 16 + r + 8 * hi;  // C layout: M = vgpr + 8*half
        int ncol = nt * 64 + j * 16 + l16;          // N = lane%16
        long idx = (long)batch * coff + (long)mrow * ldc + ncol;
        float v = acc[mi][j][r] * scale;
        if (OUT_BF16) ((__bf16*)Cv)[idx] = (__bf16)v;
        else          ((float*)Cv)[idx] = v;
      }
    }
  }
}

// ---------------- flash attention: one wave per (b, h, 16-query tile) ----------------
__global__ __launch_bounds__(32) void attn_kernel(const __bf16* __restrict__ qbf,
                                                  const __bf16* __restrict__ kbf,
                                                  const __bf16* __restrict__ vT,
                                                  __bf16* __restrict__ attbf) {
  __shared__ __align__(16) __bf16 P[16 * 32];  // probs tile, A-operand staging
  int wid = blockIdx.x;
  int qt = wid & 127;
  int h  = (wid >> 7) & 15;
  int b  = wid >> 11;
  int hk = h >> 1;  // G = 2
  int lane = threadIdx.x & 31;
  int l16 = lane & 15, hi = lane >> 4;

  // preload q fragments for all 4 d-chunks (SCALE already folded in at RoPE)
  const __bf16* qp = qbf + (((long)(b * H_ + h)) * S_ + qt * 16 + l16) * D_;
  v16bf qf[4];
#pragma unroll
  for (int dc = 0; dc < 4; ++dc) {
    v8bf lo = *(const v8bf*)(qp + dc * 32 + hi * 16);
    v8bf hi8 = *(const v8bf*)(qp + dc * 32 + 8 + hi * 16);
#pragma unroll
    for (int i = 0; i < 8; ++i) { qf[dc][i] = lo[i]; qf[dc][8 + i] = hi8[i]; }
  }

  v8f acc[8] = {};
  float mrow[8], lrow[8];
#pragma unroll
  for (int r = 0; r < 8; ++r) { mrow[r] = -1e30f; lrow[r] = 0.f; }

  const __bf16* kbase = kbf + ((long)(b * HKV_ + hk)) * S_ * D_;
  const __bf16* vbase = vT + ((long)(b * HKV_ + hk)) * D_ * (long)S_;

  int nch = qt / 2 + 1;  // 32-key chunks up to causal frontier
  for (int kc = 0; kc < nch; ++kc) {
    v8f s0 = {}, s1 = {};
#pragma unroll
    for (int dc = 0; dc < 4; ++dc) {
      v16bf b0 = *(const v16bf*)(kbase + ((long)(kc * 32 + l16)) * D_ + dc * 32 + hi * 16);
      v16bf b1 = *(const v16bf*)(kbase + ((long)(kc * 32 + 16 + l16)) * D_ + dc * 32 + hi * 16);
      s0 = wmma_bf16(qf[dc], b0, s0);
      s1 = wmma_bf16(qf[dc], b1, s1);
    }
    bool last = (kc == nch - 1);
#pragma unroll
    for (int r = 0; r < 8; ++r) {
      int prow = r + 8 * hi;
      int qrow = qt * 16 + prow;
      if (last) {
        int k0 = kc * 32 + l16;
        if (k0 > qrow)      s0[r] = -1e9f;
        if (k0 + 16 > qrow) s1[r] = -1e9f;
      }
      float t = half16_max(fmaxf(s0[r], s1[r]));
      float mn = fmaxf(mrow[r], t);
      float al = __expf(mrow[r] - mn);
      mrow[r] = mn;
      float p0 = __expf(s0[r] - mn);
      float p1 = __expf(s1[r] - mn);
      lrow[r] = lrow[r] * al + half16_sum(p0 + p1);
      P[prow * 32 + l16]      = (__bf16)p0;
      P[prow * 32 + 16 + l16] = (__bf16)p1;
#pragma unroll
      for (int dt = 0; dt < 8; ++dt) acc[dt][r] *= al;
    }
    asm volatile("s_wait_dscnt 0" ::: "memory");  // LDS in-order per wave; enforce vs compiler
    // reload probs as A-operand layout
    v8bf plo = *(const v8bf*)(&P[l16 * 32 + hi * 16]);
    v8bf phi = *(const v8bf*)(&P[l16 * 32 + 8 + hi * 16]);
    v16bf pa;
#pragma unroll
    for (int i = 0; i < 8; ++i) { pa[i] = plo[i]; pa[8 + i] = phi[i]; }
#pragma unroll
    for (int dt = 0; dt < 8; ++dt) {
      v16bf vb = *(const v16bf*)(vbase + ((long)(dt * 16 + l16)) * S_ + kc * 32 + hi * 16);
      acc[dt] = wmma_bf16(pa, vb, acc[dt]);
    }
    asm volatile("" ::: "memory");
  }

#pragma unroll
  for (int r = 0; r < 8; ++r) {
    float inv = 1.f / lrow[r];
    int row = qt * 16 + r + 8 * hi;
    long base = ((long)b * S_ + row) * (H_ * D_) + (long)h * D_;
#pragma unroll
    for (int dt = 0; dt < 8; ++dt)
      attbf[base + dt * 16 + l16] = (__bf16)(acc[dt][r] * inv);
  }
}

// ---------------- host orchestration ----------------
extern "C" void kernel_launch(void* const* d_in, const int* in_sizes, int n_in,
                              void* d_out, int out_size, void* d_ws, size_t ws_size,
                              hipStream_t stream) {
  const float* hs  = (const float*)d_in[0];
  const float* WAq = (const float*)d_in[1];
  const float* WAk = (const float*)d_in[2];
  const float* WAv = (const float*)d_in[3];
  const float* Bq  = (const float*)d_in[4];
  const float* Bk  = (const float*)d_in[5];
  const float* Bv  = (const float*)d_in[6];
  const float* Wo  = (const float*)d_in[7];
  const float* fc  = (const float*)d_in[8];
  const float* fs  = (const float*)d_in[9];
  // mask (d_in[10]) is causal -> computed in-kernel; kv_write_indices (d_in[11]) is identity.

  char* ws = (char*)d_ws;
  size_t off = 0;
  auto alloc = [&](size_t bytes) { size_t r = off; off += (bytes + 255) & ~(size_t)255; return r; };

  __bf16* Xbf  = (__bf16*)(ws + alloc((size_t)BS_ * HID_ * 2));
  __bf16* Wqb  = (__bf16*)(ws + alloc((size_t)H_ * R_ * HID_ * 2));
  __bf16* Wkb  = (__bf16*)(ws + alloc((size_t)HKV_ * R_ * HID_ * 2));
  __bf16* Wvb  = (__bf16*)(ws + alloc((size_t)HKV_ * R_ * HID_ * 2));
  __bf16* Wob  = (__bf16*)(ws + alloc((size_t)HID_ * H_ * D_ * 2));
  __bf16* BqT  = (__bf16*)(ws + alloc((size_t)H_ * D_ * R_ * 2));
  __bf16* BkT  = (__bf16*)(ws + alloc((size_t)HKV_ * D_ * R_ * 2));
  __bf16* BvT  = (__bf16*)(ws + alloc((size_t)HKV_ * D_ * R_ * 2));
  __bf16* Aqb  = (__bf16*)(ws + alloc((size_t)BS_ * H_ * R_ * 2));
  __bf16* Akb  = (__bf16*)(ws + alloc((size_t)BS_ * HKV_ * R_ * 2));
  __bf16* Avb  = (__bf16*)(ws + alloc((size_t)BS_ * HKV_ * R_ * 2));
  float*  qlin = (float*)(ws + alloc((size_t)BS_ * H_ * D_ * 4));
  float*  klin = (float*)(ws + alloc((size_t)BS_ * HKV_ * D_ * 4));
  float*  vlin = (float*)(ws + alloc((size_t)BS_ * HKV_ * D_ * 4));
  __bf16* qbf  = (__bf16*)(ws + alloc((size_t)BS_ * H_ * D_ * 2));
  __bf16* kbf  = (__bf16*)(ws + alloc((size_t)BS_ * HKV_ * D_ * 2));
  __bf16* vTb  = (__bf16*)(ws + alloc((size_t)BS_ * HKV_ * D_ * 2));
  __bf16* attb = (__bf16*)(ws + alloc((size_t)BS_ * H_ * D_ * 2));
  (void)ws_size; (void)n_in; (void)in_sizes; (void)out_size;

  auto blocks = [](long n) { return (unsigned)((n + 255) / 256); };

  // 1) fp32 -> bf16 conversions
  cvt_f32_bf16<<<blocks((long)BS_ * HID_), 256, 0, stream>>>(hs, Xbf, (long)BS_ * HID_);
  cvt_f32_bf16<<<blocks((long)H_ * R_ * HID_), 256, 0, stream>>>(WAq, Wqb, (long)H_ * R_ * HID_);
  cvt_f32_bf16<<<blocks((long)HKV_ * R_ * HID_), 256, 0, stream>>>(WAk, Wkb, (long)HKV_ * R_ * HID_);
  cvt_f32_bf16<<<blocks((long)HKV_ * R_ * HID_), 256, 0, stream>>>(WAv, Wvb, (long)HKV_ * R_ * HID_);
  cvt_f32_bf16<<<blocks((long)HID_ * H_ * D_), 256, 0, stream>>>(Wo, Wob, (long)HID_ * H_ * D_);
  transpose_rd<<<blocks((long)H_ * D_ * R_), 256, 0, stream>>>(Bq, BqT, H_);
  transpose_rd<<<blocks((long)HKV_ * D_ * R_), 256, 0, stream>>>(Bk, BkT, HKV_);
  transpose_rd<<<blocks((long)HKV_ * D_ * R_), 256, 0, stream>>>(Bv, BvT, HKV_);

  // 2) low-rank projections: A_q/A_k/A_v = X * W^T   (bf16 out)
  gemm_bt_kernel<true><<<dim3(16, 32, 1), 128, 0, stream>>>(
      Xbf, HID_, 0, Wqb, HID_, 0, Aqb, H_ * R_, 0, BS_, H_ * R_, HID_, 1.f);
  gemm_bt_kernel<true><<<dim3(8, 32, 1), 128, 0, stream>>>(
      Xbf, HID_, 0, Wkb, HID_, 0, Akb, HKV_ * R_, 0, BS_, HKV_ * R_, HID_, 1.f);
  gemm_bt_kernel<true><<<dim3(8, 32, 1), 128, 0, stream>>>(
      Xbf, HID_, 0, Wvb, HID_, 0, Avb, HKV_ * R_, 0, BS_, HKV_ * R_, HID_, 1.f);

  // 3) per-head rank-64 projections (batched over heads), scale 1/R
  gemm_bt_kernel<false><<<dim3(2, 32, H_), 128, 0, stream>>>(
      Aqb, H_ * R_, R_, BqT, R_, (long)D_ * R_, qlin, H_ * D_, D_, BS_, D_, R_, 1.f / R_);
  gemm_bt_kernel<false><<<dim3(2, 32, HKV_), 128, 0, stream>>>(
      Akb, HKV_ * R_, R_, BkT, R_, (long)D_ * R_, klin, HKV_ * D_, D_, BS_, D_, R_, 1.f / R_);
  gemm_bt_kernel<false><<<dim3(2, 32, HKV_), 128, 0, stream>>>(
      Avb, HKV_ * R_, R_, BvT, R_, (long)D_ * R_, vlin, HKV_ * D_, D_, BS_, D_, R_, 1.f / R_);

  // 4) RoPE (+ fold 1/sqrt(D) into q), layout change to head-major; v transpose to [d][s]
  rope_kernel<<<blocks((long)BS_ * H_ * 64), 256, 0, stream>>>(qlin, fc, fs, qbf, H_, SCALE_);
  rope_kernel<<<blocks((long)BS_ * HKV_ * 64), 256, 0, stream>>>(klin, fc, fs, kbf, HKV_, 1.f);
  vtrans_kernel<<<blocks((long)B_ * HKV_ * D_ * S_), 256, 0, stream>>>(vlin, vTb);

  // 5) causal flash attention (WMMA for q.k^T and P.V)
  attn_kernel<<<dim3(B_ * H_ * (S_ / 16)), 32, 0, stream>>>(qbf, kbf, vTb, attb);

  // 6) output projection: out = att * W_o^T  (f32 out -> d_out)
  gemm_bt_kernel<false><<<dim3(32, 32, 1), 128, 0, stream>>>(
      attb, H_ * D_, 0, Wob, H_ * D_, 0, (float*)d_out, HID_, 0, BS_, HID_, H_ * D_, 1.f);
}
